// Attention_21500606283937
// MI455X (gfx1250) — compile-verified, matrix operand order checked by reading, not run on hardware
//
#include <hip/hip_runtime.h>

// Flash-attention for B=4, S=4096, H=256 (fp32 in/out, additive mask).
// bf16 WMMA (v_wmma_f32_16x16x32_bf16) with f32 accumulation + online softmax.
// 1 block = 4 wave32, each wave owns a 16-row query tile; keys streamed in
// blocks of 32 through LDS. Q/K/V fragments all sourced from LDS via
// ds_load_b128 to keep VGPRs < 256; global_prefetch_b8 runs 2 blocks ahead.

#define B_ 4
#define S_ 4096
#define H_ 256

typedef __bf16 v16bf __attribute__((ext_vector_type(16)));
typedef float  v8f   __attribute__((ext_vector_type(8)));

__device__ __forceinline__ unsigned short f2bfbits(float f) {
  union { float f; unsigned u; } x; x.f = f;
  unsigned r = x.u + 0x7FFFu + ((x.u >> 16) & 1u);   // round-to-nearest-even
  return (unsigned short)(r >> 16);
}
__device__ __forceinline__ __bf16 bits2bf(unsigned short s) {
  union { unsigned short s; __bf16 b; } y; y.s = s; return y.b;
}

#define LDK 264   // bf16 row stride: >=256 and == 8 (mod 128) -> conflict-free b128 reads
#define LDP 40    // P staging stride (20-bank lane step, disjoint windows)
#define PFD 64    // prefetch distance in keys (2 key-blocks)

__global__ __launch_bounds__(128, 1)
void fa_kernel(const float* __restrict__ q, const float* __restrict__ k,
               const float* __restrict__ v, const float* __restrict__ mask,
               float* __restrict__ out) {
  __shared__ unsigned short Qs[4][16][LDK]; // per-wave Q tile: [m][h], bf16 bits
  __shared__ unsigned short Ks[32][LDK];    // K tile: [key][h], bf16 bits
  __shared__ unsigned short Vs[32][LDK];    // V tile: [key][h], bf16 bits
  __shared__ unsigned short Ps[4][16][LDP]; // per-wave P tile: [m][k], bf16 bits

  const int tid  = threadIdx.x;
  const int lane = tid & 31;
  const int wave = tid >> 5;
  const int half = lane >> 4;   // 0: lanes 0-15, 1: lanes 16-31
  const int l16  = lane & 15;

  const int nqblk = S_ / 64;
  const int b    = blockIdx.x / nqblk;
  const int qblk = (blockIdx.x % nqblk) * 64;
  const int qb   = qblk + wave * 16;

  // ---- stage Q tile (64 rows x 256 h) to LDS as bf16, cooperatively ----
#pragma unroll
  for (int p = 0; p < 32; ++p) {
    const int r  = p * 2 + (tid >> 6);   // 0..63
    const int c4 = tid & 63;             // float4 column
    const float4 qv = ((const float4*)(q + ((size_t)b * S_ + qblk + r) * H_))[c4];
    unsigned short* dst = &Qs[r >> 4][r & 15][c4 * 4];
    dst[0] = f2bfbits(qv.x); dst[1] = f2bfbits(qv.y);
    dst[2] = f2bfbits(qv.z); dst[3] = f2bfbits(qv.w);
  }

  // ---- accumulators & softmax state ----
  v8f acc[16];                 // O tile: 16 (rows) x 256 (h) f32, 16 C-tiles
#pragma unroll
  for (int j = 0; j < 16; ++j)
#pragma unroll
    for (int i = 0; i < 8; ++i) acc[j][i] = 0.0f;

  float mrow[8], lrow[8];      // per row (m = i + 8*half), replicated in 16-lane group
#pragma unroll
  for (int i = 0; i < 8; ++i) { mrow[i] = -__builtin_inff(); lrow[i] = 0.0f; }

  const float scale = 0.0625f; // 1/sqrt(256)

  // ---- hoisted per-lane base pointers (loop adds only uniform kb terms) ----
  const size_t tid64 = (size_t)tid * 64;                       // 256 B stride
  const float* kpf = k + (size_t)b * S_ * H_ + (size_t)PFD * H_ + tid64;
  const float* vpf = v + (size_t)b * S_ * H_ + (size_t)PFD * H_ + tid64;
  const float* mpf = mask + ((size_t)b * S_ + (qb + l16)) * S_ + half * 16 + PFD;
  const float* mdem = mask + ((size_t)b * S_ + qb + 8 * half) * S_ + l16;
  const float* kvb  = k + (size_t)b * S_ * H_;
  const float* vvb  = v + (size_t)b * S_ * H_;

  for (int kb = 0; kb < S_; kb += 32) {
    // ---- speculative prefetch (global_prefetch_b8), 2 key-blocks ahead ----
    if (kb < S_ - PFD) {
      __builtin_prefetch(kpf + (size_t)kb * H_, 0, 1);
      __builtin_prefetch(vpf + (size_t)kb * H_, 0, 1);
      __builtin_prefetch(mpf + kb, 0, 1);
    }

    __syncthreads();           // previous iteration's LDS consumers done

    // ---- cooperative K/V tile load: 32 keys x 256 h, f32 -> bf16 ----
#pragma unroll
    for (int p = 0; p < 16; ++p) {
      const int r  = p * 2 + (tid >> 6);   // 0..31
      const int c4 = tid & 63;             // float4 column
      const size_t row = (size_t)(kb + r) * H_;
      const float4 kv = ((const float4*)(kvb + row))[c4];
      const float4 vv = ((const float4*)(vvb + row))[c4];
      const int c = c4 * 4;
      Ks[r][c + 0] = f2bfbits(kv.x); Ks[r][c + 1] = f2bfbits(kv.y);
      Ks[r][c + 2] = f2bfbits(kv.z); Ks[r][c + 3] = f2bfbits(kv.w);
      Vs[r][c + 0] = f2bfbits(vv.x); Vs[r][c + 1] = f2bfbits(vv.y);
      Vs[r][c + 2] = f2bfbits(vv.z); Vs[r][c + 3] = f2bfbits(vv.w);
    }
    __syncthreads();

    // ---- S = Q K^T : two 16x16 f32 tiles (keys 0-15, 16-31) ----
    v8f s0, s1;
#pragma unroll
    for (int i = 0; i < 8; ++i) { s0[i] = 0.0f; s1[i] = 0.0f; }

#pragma unroll
    for (int c = 0; c < 8; ++c) {
      v16bf qa, b0, b1;        // A = Q chunk (16x32); B = K^T chunk (32h x 16keys)
#pragma unroll
      for (int e = 0; e < 16; ++e) {
        const int kd = half * 8 + (e & 7) + ((e >> 3) << 4);
        qa[e] = bits2bf(Qs[wave][l16][c * 32 + kd]);
        b0[e] = bits2bf(Ks[l16][c * 32 + kd]);
        b1[e] = bits2bf(Ks[16 + l16][c * 32 + kd]);
      }
      s0 = __builtin_amdgcn_wmma_f32_16x16x32_bf16(false, qa, false, b0,
                                                   (short)0, s0, false, false);
      s1 = __builtin_amdgcn_wmma_f32_16x16x32_bf16(false, qa, false, b1,
                                                   (short)0, s1, false, false);
    }

    // ---- scale + mask (C layout: row m = i + 8*half, col = l16) ----
#pragma unroll
    for (int i = 0; i < 8; ++i) {
      const float* mrowp = mdem + (size_t)i * S_ + kb;
      s0[i] = s0[i] * scale + mrowp[0];
      s1[i] = s1[i] * scale + mrowp[16];
    }

    // ---- online softmax over this 32-key block ----
#pragma unroll
    for (int i = 0; i < 8; ++i) {
      float mx = fmaxf(s0[i], s1[i]);
#pragma unroll
      for (int off = 8; off >= 1; off >>= 1)
        mx = fmaxf(mx, __shfl_xor(mx, off, 32));   // reduce within 16-lane group
      const float mnew  = fmaxf(mrow[i], mx);
      const float alpha = __expf(mrow[i] - mnew);
      const float p0 = __expf(s0[i] - mnew);
      const float p1 = __expf(s1[i] - mnew);
      float ps = p0 + p1;
#pragma unroll
      for (int off = 8; off >= 1; off >>= 1)
        ps += __shfl_xor(ps, off, 32);
      lrow[i] = lrow[i] * alpha + ps;
      mrow[i] = mnew;
      s0[i] = p0; s1[i] = p1;
#pragma unroll
      for (int j = 0; j < 16; ++j) acc[j][i] *= alpha;  // rescale O
    }

    // ---- stage P (C layout -> LDS row-major) for A-fragment transpose ----
#pragma unroll
    for (int i = 0; i < 8; ++i) {
      const int m = i + 8 * half;
      Ps[wave][m][l16]      = f2bfbits(s0[i]);
      Ps[wave][m][16 + l16] = f2bfbits(s1[i]);
    }
    __syncthreads();

    v16bf pf;                  // P as A-matrix 16x32
#pragma unroll
    for (int e = 0; e < 16; ++e) {
      const int kd = half * 8 + (e & 7) + ((e >> 3) << 4);
      pf[e] = bits2bf(Ps[wave][l16][kd]);
    }

    // ---- O += P V : 16 h-tiles ----
#pragma unroll
    for (int j = 0; j < 16; ++j) {
      v16bf vb;                // B = V chunk: (32 keys) x (16 h); N = l16
#pragma unroll
      for (int e = 0; e < 16; ++e) {
        const int kd = half * 8 + (e & 7) + ((e >> 3) << 4);
        vb[e] = bits2bf(Vs[kd][j * 16 + l16]);
      }
      acc[j] = __builtin_amdgcn_wmma_f32_16x16x32_bf16(false, pf, false, vb,
                                                       (short)0, acc[j], false, false);
    }
  }

  // ---- epilogue: normalize and store ----
#pragma unroll
  for (int i = 0; i < 8; ++i) {
    const float inv = 1.0f / lrow[i];
    const int qrow  = qb + i + 8 * half;
    const size_t oo = ((size_t)b * S_ + qrow) * H_;
#pragma unroll
    for (int j = 0; j < 16; ++j)
      out[oo + j * 16 + l16] = acc[j][i] * inv;
  }
}

extern "C" void kernel_launch(void* const* d_in, const int* in_sizes, int n_in,
                              void* d_out, int out_size, void* d_ws, size_t ws_size,
                              hipStream_t stream) {
  const float* q = (const float*)d_in[0];
  const float* k = (const float*)d_in[1];
  const float* v = (const float*)d_in[2];
  const float* m = (const float*)d_in[3];
  float* out = (float*)d_out;

  dim3 grid(B_ * (S_ / 64));
  dim3 block(128);
  hipLaunchKernelGGL(fa_kernel, grid, block, 0, stream, q, k, v, m, out);
}
